// RegionProposalNetwork_45380624449833
// MI455X (gfx1250) — matrix operand order repaired and checked by reading, not compiled, not run on hardware
//
#include <hip/hip_runtime.h>
#include <math.h>
#include <stdint.h>

// ---------------------------------------------------------------------------
// RPN on gfx1250: 3x3 conv (512->512, 100x100, SAME, ReLU) as implicit GEMM
// via v_wmma_f32_16x16x32_f16 (4 accumulators/wave, tap-major K ordering,
// batched A-fragment loads ahead of the wmma quartet), 1x1 heads via WMMA,
// softmax/decode, bitonic top-k sort, bitmask NMS. AMD MI455X (CDNA5, wave32).
// ---------------------------------------------------------------------------

typedef __attribute__((ext_vector_type(16))) _Float16 v16h;
typedef __attribute__((ext_vector_type(8)))  float    v8f;

#define IMGF    100
#define NPIX    10000          // 100*100
#define CIN     512
#define K1      4608           // 512*9
#define KC1     144            // 4608/32
#define NTILES  625            // 10000/16
#define CSTRIDE 10404          // 102*102 (padded plane)
#define CCHUNK  332928         // 32*10404, xpad advance per K-chunk
#define MSTRIDE (KC1 * 512)    // halves between m-tiles in wpack
#define PREK    12000
#define POSTK   2000
#define NSORT   131072
#define SUPW    375            // 12000/32
#define MINSZ   0.016f
#define IOUTHR  0.7f

// ---- padded f16 input: [512][102][102], zero border (SAME pad=1) ----------
__global__ void prep_xpad(const float* __restrict__ feat, _Float16* __restrict__ xpad) {
  int idx = blockIdx.x * blockDim.x + threadIdx.x;
  if (idx >= CIN * CSTRIDE) return;
  int c = idx / CSTRIDE;
  int r = idx - c * CSTRIDE;
  int y = r / 102, x = r - y * 102;
  float v = 0.f;
  if (y >= 1 && y <= 100 && x >= 1 && x <= 100)
    v = feat[c * NPIX + (y - 1) * IMGF + (x - 1)];
  xpad[idx] = (_Float16)v;
}

// K-offset pattern of the A fragment (contiguous, matching the B LDS read):
// lanes 0-15: halves 0..15 -> K 0..15
// lanes 16-31: halves 0..15 -> K 16..31
__device__ __forceinline__ int koff_of(int lane, int i) {
  int base = i;
  return (lane < 16) ? base : base + 16;
}

// ---- pack w_inter (512,512,3,3) into per-lane contiguous A fragments ------
// K ordering is tap-major: K = j*512 + c  (j = kh*3+kw tap, c = in-channel),
// so each 32-wide K-chunk sits inside one tap. wpack[mt(32)][kc(144)][lane][i]
__global__ void prep_wpack1(const float* __restrict__ w, _Float16* __restrict__ wp) {
  int idx = blockIdx.x * blockDim.x + threadIdx.x;
  if (idx >= 32 * KC1 * 32 * 16) return;
  int i    = idx & 15;
  int lane = (idx >> 4) & 31;
  int rest = idx >> 9;
  int kc   = rest % KC1;
  int mt   = rest / KC1;
  int M = mt * 16 + (lane & 15);
  int K = kc * 32 + koff_of(lane, i);
  int j = K >> 9, c = K & 511;                 // tap-major decode (no div!)
  wp[idx] = (_Float16)w[(size_t)M * K1 + c * 9 + j];
}

// ---- pack heads: rows 0..35 = w_reg, 36..53 = w_cls, 54..63 = zero --------
// wpack2[mt2(4)][kc(16)][lane(32)][i(16)]
__global__ void prep_wpack2(const float* __restrict__ wreg, const float* __restrict__ wcls,
                            _Float16* __restrict__ wp) {
  int idx = blockIdx.x * blockDim.x + threadIdx.x;
  if (idx >= 4 * 16 * 32 * 16) return;
  int i    = idx & 15;
  int lane = (idx >> 4) & 31;
  int rest = idx >> 9;
  int kc   = rest & 15;
  int mt   = rest >> 4;
  int M = mt * 16 + (lane & 15);
  int K = kc * 32 + koff_of(lane, i);   // K == input channel for 1x1 conv
  float v = 0.f;
  if (M < 36)       v = wreg[M * CIN + K];
  else if (M < 54)  v = wcls[(M - 36) * CIN + K];
  wp[idx] = (_Float16)v;
}

// ---- conv1: implicit GEMM M=512 N=10000 K=4608, WMMA f16->f32, ReLU -------
// One block per 16-pixel tile covers all 512 out-channels: 8 waves x 4 m-tiles
// each. Per staged chunk: 8 clause-batched b128 A-loads, then 4 back-to-back
// wmma sharing one B fragment.
__global__ __launch_bounds__(256) void conv1_wmma(
    const _Float16* __restrict__ xpad, const _Float16* __restrict__ wpack,
    const float* __restrict__ b_inter, _Float16* __restrict__ x16) {
  __shared__ _Float16 Bs[16 * 48];          // [n][k] rows padded to 48 halves (96B)
  const int wave  = threadIdx.x >> 5;
  const int lane  = threadIdx.x & 31;
  const int pbase = blockIdx.x * 16;

  // staging invariants: this thread feeds Bs[n0][k0] and Bs[n0][k0+16]
  const int k0 = threadIdx.x >> 4;          // 0..15
  const int n0 = threadIdx.x & 15;
  const int p0 = pbase + n0;
  const int ph = p0 / IMGF, pw = p0 - ph * IMGF;
  const int base_a = k0 * CSTRIDE + ph * 102 + pw;
  const int base_b = (k0 + 16) * CSTRIDE + ph * 102 + pw;
  _Float16* const bsa = &Bs[n0 * 48 + k0];
  _Float16* const bsb = &Bs[n0 * 48 + k0 + 16];
  const _Float16* const bsrd = Bs + (lane & 15) * 48 + (lane >> 4) * 16;

  // single running A pointer; 4 m-tiles addressed by constant 144KB offsets
  const _Float16* ap = wpack + ((size_t)(wave * 4) * MSTRIDE) + (size_t)lane * 16;

  v8f acc[4] = {};
  for (int j = 0; j < 9; ++j) {             // 3x3 tap (uniform scalar per iter)
    const int dh = j / 3, dw = j - dh * 3;
    int off_a = base_a + dh * 102 + dw;
    int off_b = base_b + dh * 102 + dw;
    for (int cc = 0; cc < 16; ++cc) {       // 16 channel-chunks per tap
      *bsa = xpad[off_a];
      *bsb = xpad[off_b];
      __builtin_prefetch((const void*)(xpad + off_a + CCHUNK), 0, 1);
      __syncthreads();
      // load all fragments first (one clause), then 4 wmma back-to-back
      v16h af0 = *(const v16h*)(ap);
      v16h af1 = *(const v16h*)(ap + 1 * MSTRIDE);
      v16h af2 = *(const v16h*)(ap + 2 * MSTRIDE);
      v16h af3 = *(const v16h*)(ap + 3 * MSTRIDE);
      v16h b   = *(const v16h*)bsrd;
      acc[0] = __builtin_amdgcn_wmma_f32_16x16x32_f16(false, af0, false, b,
                                                      (short)0, acc[0], false, false);
      acc[1] = __builtin_amdgcn_wmma_f32_16x16x32_f16(false, af1, false, b,
                                                      (short)0, acc[1], false, false);
      acc[2] = __builtin_amdgcn_wmma_f32_16x16x32_f16(false, af2, false, b,
                                                      (short)0, acc[2], false, false);
      acc[3] = __builtin_amdgcn_wmma_f32_16x16x32_f16(false, af3, false, b,
                                                      (short)0, acc[3], false, false);
      ap += 512;                            // next K-chunk fragment
      __syncthreads();
      off_a += CCHUNK;
      off_b += CCHUNK;
    }
  }
  // C/D layout: VGPR r -> M = r (+8 for lanes 16..31), N = lane%16
  const int p = pbase + (lane & 15);
  const int rowoff = (lane >> 4) ? 8 : 0;
  #pragma unroll
  for (int t = 0; t < 4; ++t) {
    const int mb = (wave * 4 + t) * 16 + rowoff;
    #pragma unroll
    for (int r = 0; r < 8; ++r) {
      int m = mb + r;
      float v = acc[t][r] + b_inter[m];
      v = v > 0.f ? v : 0.f;                // ReLU
      x16[(size_t)m * NPIX + p] = (_Float16)v;
    }
  }
}

// ---- conv2: 1x1 heads, M=64(pad) N=10000 K=512 ----------------------------
__global__ __launch_bounds__(128) void conv2_wmma(
    const _Float16* __restrict__ x16, const _Float16* __restrict__ wpack2,
    const float* __restrict__ b_reg, const float* __restrict__ b_cls,
    float* __restrict__ yconv) {
  __shared__ _Float16 Bs[16 * 48];
  const int wave = threadIdx.x >> 5;        // m-subtile 0..3
  const int lane = threadIdx.x & 31;
  const int pbase = blockIdx.x * 16;

  v8f acc = {};
  for (int kc = 0; kc < 16; ++kc) {
    #pragma unroll
    for (int e = 0; e < 4; ++e) {
      int lin = threadIdx.x + e * 128;
      int k = lin >> 4, n = lin & 15;
      Bs[n * 48 + k] = x16[(size_t)(kc * 32 + k) * NPIX + pbase + n];
    }
    __syncthreads();
    v16h a = *(const v16h*)(wpack2 + ((size_t)(wave * 16 + kc) * 32 + lane) * 16);
    v16h b = *(const v16h*)(Bs + (lane & 15) * 48 + (lane >> 4) * 16);
    acc = __builtin_amdgcn_wmma_f32_16x16x32_f16(false, a, false, b,
                                                 (short)0, acc, false, false);
    __syncthreads();
  }
  const int p  = pbase + (lane & 15);
  const int mb = wave * 16 + ((lane >> 4) ? 8 : 0);
  #pragma unroll
  for (int r = 0; r < 8; ++r) {
    int m = mb + r;
    if (m < 54) {
      float bias = (m < 36) ? b_reg[m] : b_cls[m - 36];
      yconv[(size_t)m * NPIX + p] = acc[r] + bias;
    }
  }
}

__device__ __forceinline__ uint32_t ord_of(float f) {
  uint32_t b = __float_as_uint(f);
  return (b & 0x80000000u) ? ~b : (b | 0x80000000u);   // monotone float->uint
}

// ---- postprocess: outputs, softmax fg, decode+clip, score, sort keys ------
__global__ void postprocess(const float* __restrict__ yconv, const float* __restrict__ anchor,
                            float* __restrict__ out_reg, float* __restrict__ out_cls,
                            float* __restrict__ out_anchor, float* __restrict__ roi_all,
                            unsigned long long* __restrict__ key64) {
  int i = blockIdx.x * blockDim.x + threadIdx.x;
  if (i >= NSORT) return;
  if (i >= 90000) { key64[i] = ~0ull; return; }        // pad entries sort last
  int p = i / 9, a = i - p * 9;

  float t[4];
  #pragma unroll
  for (int d = 0; d < 4; ++d) {
    t[d] = yconv[(size_t)(a * 4 + d) * NPIX + p];
    out_reg[(size_t)i * 4 + d] = t[d];
  }
  float c0 = yconv[(size_t)(36 + a * 2 + 0) * NPIX + p];
  float c1 = yconv[(size_t)(36 + a * 2 + 1) * NPIX + p];
  out_cls[(size_t)i * 2 + 0] = c0;
  out_cls[(size_t)i * 2 + 1] = c1;
  float fg = 1.f / (1.f + expf(c0 - c1));              // softmax[...,1]

  float ax1 = anchor[i * 4 + 0], ay1 = anchor[i * 4 + 1];
  float ax2 = anchor[i * 4 + 2], ay2 = anchor[i * 4 + 3];
  out_anchor[i * 4 + 0] = ax1; out_anchor[i * 4 + 1] = ay1;
  out_anchor[i * 4 + 2] = ax2; out_anchor[i * 4 + 3] = ay2;

  float aw = ax2 - ax1, ah = ay2 - ay1;
  float acx = (ax1 + ax2) * 0.5f, acy = (ay1 + ay2) * 0.5f;
  float cx = t[0] * aw + acx, cy = t[1] * ah + acy;
  float bw = expf(t[2]) * aw, bh = expf(t[3]) * ah;
  float x1 = fminf(fmaxf(cx - 0.5f * bw, 0.f), 1.f);
  float y1 = fminf(fmaxf(cy - 0.5f * bh, 0.f), 1.f);
  float x2 = fminf(fmaxf(cx + 0.5f * bw, 0.f), 1.f);
  float y2 = fminf(fmaxf(cy + 0.5f * bh, 0.f), 1.f);
  roi_all[(size_t)i * 4 + 0] = x1; roi_all[(size_t)i * 4 + 1] = y1;
  roi_all[(size_t)i * 4 + 2] = x2; roi_all[(size_t)i * 4 + 3] = y2;

  bool ok = ((y2 - y1) >= MINSZ) && ((x2 - x1) >= MINSZ);
  float score = ok ? fg : -INFINITY;
  // key: score descending, index ascending on ties (matches lax.top_k)
  key64[i] = ((unsigned long long)(~ord_of(score)) << 32) | (unsigned)i;
}

// ---- bitonic sort pass (ascending on 64-bit key) --------------------------
__global__ void bitonic_pass(unsigned long long* __restrict__ key, unsigned j, unsigned k) {
  unsigned i = blockIdx.x * blockDim.x + threadIdx.x;
  unsigned ixj = i ^ j;
  if (ixj > i) {
    unsigned long long a = key[i], b = key[ixj];
    bool up = ((i & k) == 0);
    if ((a > b) == up) { key[i] = b; key[ixj] = a; }
  }
}

// ---- gather top-12000 boxes, areas, validity ------------------------------
__global__ void gather12k(const unsigned long long* __restrict__ key64,
                          const float* __restrict__ roi_all,
                          float* __restrict__ boxes, float* __restrict__ areas,
                          int* __restrict__ valid) {
  int t = blockIdx.x * blockDim.x + threadIdx.x;
  if (t >= PREK) return;
  unsigned long long e = key64[t];
  unsigned idx = (unsigned)e;
  unsigned u = ~(unsigned)(e >> 32);                   // ord(score)
  int v = (u > 0x007FFFFFu) ? 1 : 0;                   // score > -inf
  float x1 = 0.f, y1 = 0.f, x2 = 0.f, y2 = 0.f;
  if (idx < 90000) {
    x1 = roi_all[(size_t)idx * 4 + 0]; y1 = roi_all[(size_t)idx * 4 + 1];
    x2 = roi_all[(size_t)idx * 4 + 2]; y2 = roi_all[(size_t)idx * 4 + 3];
  }
  boxes[t * 4 + 0] = x1; boxes[t * 4 + 1] = y1;
  boxes[t * 4 + 2] = x2; boxes[t * 4 + 3] = y2;
  areas[t] = (x2 - x1) * (y2 - y1);
  valid[t] = v;
}

// ---- suppression bitmask: row i, bit j -> (j>i && IOU>thr) ----------------
__global__ __launch_bounds__(384) void supp_matrix(const float* __restrict__ boxes,
                                                   const float* __restrict__ areas,
                                                   uint32_t* __restrict__ mat) {
  int row = blockIdx.x;
  int w = threadIdx.x;
  if (w >= SUPW) return;
  float rx1 = boxes[row * 4 + 0], ry1 = boxes[row * 4 + 1];
  float rx2 = boxes[row * 4 + 2], ry2 = boxes[row * 4 + 3];
  float ra = areas[row];
  uint32_t word = 0;
  #pragma unroll 4
  for (int b = 0; b < 32; ++b) {
    int col = w * 32 + b;                              // 375*32 == 12000 exactly
    float xx1 = fmaxf(rx1, boxes[col * 4 + 0]);
    float yy1 = fmaxf(ry1, boxes[col * 4 + 1]);
    float xx2 = fminf(rx2, boxes[col * 4 + 2]);
    float yy2 = fminf(ry2, boxes[col * 4 + 3]);
    float inter = fmaxf(xx2 - xx1, 0.f) * fmaxf(yy2 - yy1, 0.f);
    float iou = inter / (ra + areas[col] - inter + 1e-12f);
    if (col > row && iou > IOUTHR) word |= (1u << b);
  }
  mat[(size_t)row * SUPW + w] = word;
}

// ---- greedy NMS scan + compaction into ROI output (single workgroup) ------
__global__ __launch_bounds__(384) void nms_scan(const uint32_t* __restrict__ mat,
                                                const int* __restrict__ valid,
                                                const float* __restrict__ boxes,
                                                float* __restrict__ out_roi) {
  __shared__ uint32_t supp[SUPW];
  const int w = threadIdx.x;
  if (w < SUPW) {
    uint32_t s = 0;
    for (int b = 0; b < 32; ++b)
      if (!valid[w * 32 + b]) s |= (1u << b);          // supp init = !valid
    supp[w] = s;
  }
  for (int z = w; z < POSTK * 4; z += 384) out_roi[z] = 0.f;
  __syncthreads();

  for (int i = 0; i < PREK; ++i) {
    uint32_t fw = supp[i >> 5];                        // uniform read
    __syncthreads();
    if (!((fw >> (i & 31)) & 1u)) {                    // box i survives -> suppress
      if (w < SUPW) supp[w] |= mat[(size_t)i * SUPW + w];
    }
    __syncthreads();
  }
  if (w == 0) {
    int rank = 0;
    for (int i = 0; i < PREK && rank < POSTK; ++i) {
      bool kept = !((supp[i >> 5] >> (i & 31)) & 1u) && valid[i];
      if (kept) {
        out_roi[rank * 4 + 0] = boxes[i * 4 + 0];
        out_roi[rank * 4 + 1] = boxes[i * 4 + 1];
        out_roi[rank * 4 + 2] = boxes[i * 4 + 2];
        out_roi[rank * 4 + 3] = boxes[i * 4 + 3];
        ++rank;
      }
    }
  }
}

// ---------------------------------------------------------------------------
static inline char* bump(char*& p, size_t bytes) {
  char* r = p;
  p += (bytes + 255) & ~(size_t)255;
  return r;
}

extern "C" void kernel_launch(void* const* d_in, const int* in_sizes, int n_in,
                              void* d_out, int out_size, void* d_ws, size_t ws_size,
                              hipStream_t stream) {
  (void)in_sizes; (void)n_in; (void)out_size; (void)ws_size;
  const float* features = (const float*)d_in[0];
  const float* w_inter  = (const float*)d_in[1];
  const float* b_inter  = (const float*)d_in[2];
  const float* w_reg    = (const float*)d_in[3];
  const float* b_reg    = (const float*)d_in[4];
  const float* w_cls    = (const float*)d_in[5];
  const float* b_cls    = (const float*)d_in[6];
  const float* anchor   = (const float*)d_in[7];

  float* out        = (float*)d_out;
  float* out_reg    = out;                 // 90000*4
  float* out_cls    = out + 360000;        // 90000*2
  float* out_roi    = out + 540000;        // 2000*4
  float* out_anchor = out + 548000;        // 90000*4

  char* wsp = (char*)d_ws;
  _Float16* xpad   = (_Float16*)bump(wsp, (size_t)CIN * CSTRIDE * 2);
  _Float16* wpack  = (_Float16*)bump(wsp, (size_t)32 * KC1 * 32 * 16 * 2);
  _Float16* wpack2 = (_Float16*)bump(wsp, (size_t)4 * 16 * 32 * 16 * 2);
  _Float16* x16    = (_Float16*)bump(wsp, (size_t)CIN * NPIX * 2);
  float*    yconv  = (float*)   bump(wsp, (size_t)54 * NPIX * 4);
  float*    roiall = (float*)   bump(wsp, (size_t)90000 * 4 * 4);
  unsigned long long* key64 = (unsigned long long*)bump(wsp, (size_t)NSORT * 8);
  float*    boxes  = (float*)   bump(wsp, (size_t)PREK * 4 * 4);
  float*    areas  = (float*)   bump(wsp, (size_t)PREK * 4);
  int*      valid  = (int*)     bump(wsp, (size_t)PREK * 4);
  uint32_t* mat    = (uint32_t*)bump(wsp, (size_t)PREK * SUPW * 4);

  // 1. prep: padded f16 input + packed WMMA A-fragments
  prep_xpad  <<<(CIN * CSTRIDE + 255) / 256, 256, 0, stream>>>(features, xpad);
  prep_wpack1<<<(32 * KC1 * 32 * 16 + 255) / 256, 256, 0, stream>>>(w_inter, wpack);
  prep_wpack2<<<(4 * 16 * 32 * 16 + 255) / 256, 256, 0, stream>>>(w_reg, w_cls, wpack2);

  // 2. 3x3 conv as implicit GEMM (WMMA) + ReLU -> f16 activations
  conv1_wmma<<<NTILES, 256, 0, stream>>>(xpad, wpack, b_inter, x16);

  // 3. 1x1 reg/cls heads (WMMA), bias added
  conv2_wmma<<<NTILES, 128, 0, stream>>>(x16, wpack2, b_reg, b_cls, yconv);

  // 4. outputs + softmax + decode/clip/filter + sort keys
  postprocess<<<NSORT / 256, 256, 0, stream>>>(yconv, anchor, out_reg, out_cls,
                                               out_anchor, roiall, key64);

  // 5. bitonic sort (score desc, index asc) — 153 graph-capturable passes
  for (unsigned k = 2; k <= NSORT; k <<= 1)
    for (unsigned j = k >> 1; j > 0; j >>= 1)
      bitonic_pass<<<NSORT / 256, 256, 0, stream>>>(key64, j, k);

  // 6. top-12000 gather, IOU suppression matrix, greedy scan + compaction
  gather12k  <<<(PREK + 255) / 256, 256, 0, stream>>>(key64, roiall, boxes, areas, valid);
  supp_matrix<<<PREK, 384, 0, stream>>>(boxes, areas, mat);
  nms_scan   <<<1, 384, 0, stream>>>(mat, valid, boxes, out_roi);
}
